// CombinedGraphReadout_6571299963462
// MI455X (gfx1250) — compile-verified, matrix operand order checked by reading, not run on hardware
//
#include <hip/hip_runtime.h>
#include <hip/hip_bf16.h>

typedef __attribute__((ext_vector_type(16))) __bf16 v16bf;
typedef __attribute__((ext_vector_type(8)))  __bf16 v8bf;
typedef __attribute__((ext_vector_type(8)))  float  v8f;

#define NODE_DIM 256
#define HIDC     256
#define NHEADS   8
#define OUTD     512

// ---------------- WMMA helpers ----------------

__device__ __forceinline__ v8f wmma_bf16(v16bf a, v16bf b, v8f c) {
    // 8 args: (neg_a, A, neg_b, B, c_mod, C, reuse_a, reuse_b)
    return __builtin_amdgcn_wmma_f32_16x16x32_bf16(false, a, false, b, (short)0, c, false, false);
}

// A-fragment (16x32 bf16) loader from an f32 row pointer (row m of this lane).
// Layout (ISA 7.12.2, 16-bit A): lane<16 -> K={kf*32+0..7, kf*32+16..23},
// lane>=16 -> K={kf*32+8..15, kf*32+24..31}. hl = lane>>4.
__device__ __forceinline__ v16bf a_frag_f32(const float* __restrict__ row, int kf, int hl) {
    const float* p1 = row + kf * 32 + hl * 8;
    const float* p2 = p1 + 16;
    v16bf a;
#pragma unroll
    for (int i = 0; i < 8; ++i) a[i] = (__bf16)p1[i];
#pragma unroll
    for (int i = 0; i < 8; ++i) a[8 + i] = (__bf16)p2[i];
    return a;
}

__device__ __forceinline__ v16bf a_frag_bf16(const __bf16* row, int kf, int hl) {
    const __bf16* p1 = row + kf * 32 + hl * 8;
    const __bf16* p2 = p1 + 16;
    v16bf a;
#pragma unroll
    for (int i = 0; i < 8; ++i) a[i] = p1[i];
#pragma unroll
    for (int i = 0; i < 8; ++i) a[8 + i] = p2[i];
    return a;
}

// B fragment: prepacked so each lane reads 16 contiguous bf16 (32B, coalesced 1KB/wave).
__device__ __forceinline__ v16bf b_frag(const __bf16* __restrict__ packed, int kfrags,
                                        int nt, int kf, int lane) {
    const __bf16* p = packed + (((size_t)nt * kfrags + kf) << 9) + (lane << 4);
    return *(const v16bf*)p;
}

// Wave-local LDS ordering (per-wave private LDS tile; no cross-wave sharing).
__device__ __forceinline__ void wave_sync_lds() {
    __builtin_amdgcn_wave_barrier();
    asm volatile("s_wait_dscnt 0" ::: "memory");
    __builtin_amdgcn_wave_barrier();
}

// ---------------- kernel 0: weight repack f32 -> WMMA-B bf16 fragments ----------------

__global__ void k_pack_b(const float* __restrict__ src, __bf16* __restrict__ dst,
                         int K, int Ncols, int realN) {
    int kfrags = K >> 5;
    int total = kfrags * (Ncols >> 4) * 512;
    for (int idx = blockIdx.x * blockDim.x + threadIdx.x; idx < total;
         idx += gridDim.x * blockDim.x) {
        int f    = idx >> 9;
        int lane = (idx >> 4) & 31;
        int i    = idx & 15;
        int nt = f / kfrags;
        int kf = f - nt * kfrags;
        int n  = nt * 16 + (lane & 15);
        int hl = lane >> 4;
        int k  = kf * 32 + ((i < 8) ? (hl * 8 + i) : (16 + hl * 8 + (i - 8)));
        float v = (n < realN) ? src[(size_t)k * realN + n] : 0.0f;
        dst[idx] = (__bf16)v;
    }
}

// ---------------- kernel 1: segment bounds (ids sorted) ----------------

__global__ void k_bounds(const int* __restrict__ seg, int N, int G, int* __restrict__ bounds) {
    int g = blockIdx.x * blockDim.x + threadIdx.x;
    if (g > G) return;
    int lo = 0, hi = N;
    while (lo < hi) {
        int mid = (lo + hi) >> 1;
        if (seg[mid] < g) lo = mid + 1; else hi = mid;
    }
    bounds[g] = lo;
}

// ---------------- kernel 2: fused per-node MLPs (WMMA) ----------------
// One wave = 32 nodes (2 row-subtiles) so every register-resident B fragment
// feeds 2 WMMAs (halves L2 B-fragment traffic vs 16 nodes/wave).
// For both poolers:
//   scores = relu(x@W1+b1)@W2+b2        -> sc_[N,8] f32
//   values = relu(x@V1+c1)@V2+c2        -> vT_[256,N] bf16 (transposed)

__global__ __launch_bounds__(128) void k_node_mlps(
    const float* __restrict__ x, int N,
    const __bf16* __restrict__ pwA_s1, const float* __restrict__ bA_s1,
    const __bf16* __restrict__ pwA_s2, const float* __restrict__ bA_s2,
    const __bf16* __restrict__ pwA_v1, const float* __restrict__ bA_v1,
    const __bf16* __restrict__ pwA_v2, const float* __restrict__ bA_v2,
    const __bf16* __restrict__ pwB_s1, const float* __restrict__ bB_s1,
    const __bf16* __restrict__ pwB_s2, const float* __restrict__ bB_s2,
    const __bf16* __restrict__ pwB_v1, const float* __restrict__ bB_v1,
    const __bf16* __restrict__ pwB_v2, const float* __restrict__ bB_v2,
    float* __restrict__ scA, float* __restrict__ scB,
    __bf16* __restrict__ vTA, __bf16* __restrict__ vTB) {

    __shared__ __bf16 hbuf[4][32][HIDC];   // 64 KB: per-wave 32x256 activation tile

    const int lane = threadIdx.x & 31;
    const int w    = threadIdx.x >> 5;
    const int tile = blockIdx.x * 4 + w;          // 32-node tile index
    if (tile * 32 >= N) return;
    const int mbase = tile * 32;
    const int mrow  = lane & 15;
    const int hl    = lane >> 4;

    // Load the 32x256 x tile once as 2x8 A-fragments (bf16).
    v16bf xfr[2][8];
#pragma unroll
    for (int s = 0; s < 2; ++s) {
        int r = mbase + s * 16 + mrow;
        if (r >= N) r = N - 1;
        const float* xrow = x + (size_t)r * NODE_DIM;
#pragma unroll
        for (int kf = 0; kf < 8; ++kf) xfr[s][kf] = a_frag_f32(xrow, kf, hl);
    }

    const __bf16* W1[2] = {pwA_s1, pwB_s1};
    const __bf16* W2[2] = {pwA_s2, pwB_s2};
    const __bf16* V1[2] = {pwA_v1, pwB_v1};
    const __bf16* V2[2] = {pwA_v2, pwB_v2};
    const float*  B1[2] = {bA_s1, bB_s1};
    const float*  B2[2] = {bA_s2, bB_s2};
    const float*  C1[2] = {bA_v1, bB_v1};
    const float*  C2[2] = {bA_v2, bB_v2};
    float*  SC[2] = {scA, scB};
    __bf16* VT[2] = {vTA, vTB};

    __bf16 (*hrow)[HIDC] = hbuf[w];

#pragma unroll
    for (int p = 0; p < 2; ++p) {
        // ---- score layer 1: h = relu(x @ W1 + b1), 32x256 out ----
        for (int nt = 0; nt < 16; ++nt) {
            v16bf bfr[8];
#pragma unroll
            for (int kf = 0; kf < 8; ++kf) bfr[kf] = b_frag(W1[p], 8, nt, kf, lane);
            v8f acc0 = {}, acc1 = {};
#pragma unroll
            for (int kf = 0; kf < 8; ++kf) {
                acc0 = wmma_bf16(xfr[0][kf], bfr[kf], acc0);
                acc1 = wmma_bf16(xfr[1][kf], bfr[kf], acc1);
            }
            float bias = B1[p][nt * 16 + mrow];
#pragma unroll
            for (int r = 0; r < 8; ++r) {
                hrow[r + 8 * hl][nt * 16 + mrow]      = (__bf16)fmaxf(acc0[r] + bias, 0.0f);
                hrow[16 + r + 8 * hl][nt * 16 + mrow] = (__bf16)fmaxf(acc1[r] + bias, 0.0f);
            }
        }
        wave_sync_lds();
        v16bf hfr[2][8];
#pragma unroll
        for (int s = 0; s < 2; ++s)
#pragma unroll
            for (int kf = 0; kf < 8; ++kf)
                hfr[s][kf] = a_frag_bf16(&hrow[s * 16 + mrow][0], kf, hl);

        // ---- score layer 2: 256 -> 8 (padded to 16 cols) ----
        {
            v16bf bfr[8];
#pragma unroll
            for (int kf = 0; kf < 8; ++kf) bfr[kf] = b_frag(W2[p], 8, 0, kf, lane);
            v8f acc0 = {}, acc1 = {};
#pragma unroll
            for (int kf = 0; kf < 8; ++kf) {
                acc0 = wmma_bf16(hfr[0][kf], bfr[kf], acc0);
                acc1 = wmma_bf16(hfr[1][kf], bfr[kf], acc1);
            }
            if (mrow < NHEADS) {
                float bias = B2[p][mrow];
#pragma unroll
                for (int r = 0; r < 8; ++r) {
                    int n0 = mbase + r + 8 * hl;
                    int n1 = n0 + 16;
                    if (n0 < N) SC[p][(size_t)n0 * NHEADS + mrow] = acc0[r] + bias;
                    if (n1 < N) SC[p][(size_t)n1 * NHEADS + mrow] = acc1[r] + bias;
                }
            }
        }
        wave_sync_lds();

        // ---- value layer 1: h = relu(x @ V1 + c1) ----
        for (int nt = 0; nt < 16; ++nt) {
            v16bf bfr[8];
#pragma unroll
            for (int kf = 0; kf < 8; ++kf) bfr[kf] = b_frag(V1[p], 8, nt, kf, lane);
            v8f acc0 = {}, acc1 = {};
#pragma unroll
            for (int kf = 0; kf < 8; ++kf) {
                acc0 = wmma_bf16(xfr[0][kf], bfr[kf], acc0);
                acc1 = wmma_bf16(xfr[1][kf], bfr[kf], acc1);
            }
            float bias = C1[p][nt * 16 + mrow];
#pragma unroll
            for (int r = 0; r < 8; ++r) {
                hrow[r + 8 * hl][nt * 16 + mrow]      = (__bf16)fmaxf(acc0[r] + bias, 0.0f);
                hrow[16 + r + 8 * hl][nt * 16 + mrow] = (__bf16)fmaxf(acc1[r] + bias, 0.0f);
            }
        }
        wave_sync_lds();
#pragma unroll
        for (int s = 0; s < 2; ++s)
#pragma unroll
            for (int kf = 0; kf < 8; ++kf)
                hfr[s][kf] = a_frag_bf16(&hrow[s * 16 + mrow][0], kf, hl);

        // ---- value layer 2: v = h @ V2 + c2, stored transposed [256][N] bf16 ----
        for (int nt = 0; nt < 16; ++nt) {
            v16bf bfr[8];
#pragma unroll
            for (int kf = 0; kf < 8; ++kf) bfr[kf] = b_frag(V2[p], 8, nt, kf, lane);
            v8f acc0 = {}, acc1 = {};
#pragma unroll
            for (int kf = 0; kf < 8; ++kf) {
                acc0 = wmma_bf16(hfr[0][kf], bfr[kf], acc0);
                acc1 = wmma_bf16(hfr[1][kf], bfr[kf], acc1);
            }
            float bias = C2[p][nt * 16 + mrow];
            size_t col = (size_t)(nt * 16 + mrow);
            if (mbase + 32 <= N) {
                v8bf tv0, tv1;
#pragma unroll
                for (int r = 0; r < 8; ++r) {
                    tv0[r] = (__bf16)(acc0[r] + bias);
                    tv1[r] = (__bf16)(acc1[r] + bias);
                }
                *(v8bf*)(VT[p] + col * N + mbase + 8 * hl)      = tv0;  // 16B/lane
                *(v8bf*)(VT[p] + col * N + mbase + 16 + 8 * hl) = tv1;
            } else {
#pragma unroll
                for (int r = 0; r < 8; ++r) {
                    int n0 = mbase + r + 8 * hl;
                    int n1 = n0 + 16;
                    if (n0 < N) VT[p][col * N + n0] = (__bf16)(acc0[r] + bias);
                    if (n1 < N) VT[p][col * N + n1] = (__bf16)(acc1[r] + bias);
                }
            }
        }
        wave_sync_lds();
    }
}

// ---------------- kernel 3: per-graph softmax stats + segment pools ----------------
// One block (256 threads) per graph; thread t owns channel t.

__global__ __launch_bounds__(256) void k_segment_pool(
    const float* __restrict__ x, const int* __restrict__ bounds, int N,
    const float* __restrict__ sc_wm, const float* __restrict__ sc_ws,
    const __bf16* __restrict__ vT_wm, const __bf16* __restrict__ vT_ws,
    float* __restrict__ pgw, float* __restrict__ pgs, float* __restrict__ pgm) {

    __shared__ float red[256];
    __shared__ float mh[NHEADS];
    __shared__ float sh[NHEADS];

    const int g = blockIdx.x;
    const int t = threadIdx.x;
    const int start = bounds[g];
    const int end   = bounds[g + 1];

    // scatter-softmax stats for wm scores: 32 workers per head
    const int h  = t & 7;
    const int li = t >> 3;
    float mx = -3.402823466e38f;
    for (int n = start + li; n < end; n += 32)
        mx = fmaxf(mx, sc_wm[(size_t)n * NHEADS + h]);
    red[t] = mx;
    __syncthreads();
    for (int s = 16; s > 0; s >>= 1) {
        if (li < s) red[t] = fmaxf(red[t], red[t + s * 8]);
        __syncthreads();
    }
    if (t < NHEADS) mh[t] = red[t];
    __syncthreads();
    float m = mh[h];
    float sm = 0.0f;
    for (int n = start + li; n < end; n += 32)
        sm += __expf(sc_wm[(size_t)n * NHEADS + h] - m);
    red[t] = sm;
    __syncthreads();
    for (int s = 16; s > 0; s >>= 1) {
        if (li < s) red[t] += red[t + s * 8];
        __syncthreads();
    }
    if (t < NHEADS) sh[t] = (red[t] > 0.0f) ? 1.0f / red[t] : 0.0f;
    __syncthreads();

    // main accumulation: channel c = t, head = c/32
    const int c  = t;
    const int h2 = c >> 5;
    const float mv = mh[h2];
    const float si = sh[h2];
    const __bf16* vwc = vT_wm + (size_t)c * N;
    const __bf16* vsc = vT_ws + (size_t)c * N;
    float aw = 0.0f, as_ = 0.0f, am = -3.402823466e38f;
    for (int n = start; n < end; ++n) {
        float xv = x[(size_t)n * NODE_DIM + c];          // fully coalesced per block
        am = fmaxf(am, xv);
        float e  = __expf(sc_wm[(size_t)n * NHEADS + h2] - mv) * si;
        float sg = 1.0f / (1.0f + __expf(-sc_ws[(size_t)n * NHEADS + h2]));
        aw  += e  * (float)vwc[n];
        as_ += sg * (float)vsc[n];
    }
    pgw[(size_t)g * HIDC + c] = aw;
    pgs[(size_t)g * HIDC + c] = as_;
    pgm[(size_t)g * HIDC + c] = (end > start) ? am : 0.0f;
}

// ---------------- kernel 4: per-graph GEMMs (WMMA) ----------------
// out = relu([pgw@Cw | pgs@Cs | pgm@Cm]) @ F     (16 graphs per block)

__global__ __launch_bounds__(256) void k_graph_out(
    const float* __restrict__ pgw, const float* __restrict__ pgs, const float* __restrict__ pgm,
    const __bf16* __restrict__ p_cw, const __bf16* __restrict__ p_cs, const __bf16* __restrict__ p_cm,
    const __bf16* __restrict__ p_fin,
    float* __restrict__ out, int G) {

    __shared__ __bf16 raw[16][3 * OUTD];   // 16 x 1536 bf16 = 48 KB

    const int lane  = threadIdx.x & 31;
    const int w     = threadIdx.x >> 5;
    const int gbase = blockIdx.x * 16;
    const int mrow  = lane & 15;
    const int hl    = lane >> 4;

    const float*  srcs[3] = {pgw, pgs, pgm};
    const __bf16* pk[3]   = {p_cw, p_cs, p_cm};

    // raw has 96 n-tiles of 16 cols; 12 per wave
    for (int j = w; j < 96; j += 8) {
        int which = j >> 5;
        int nt    = j & 31;
        int gr    = gbase + mrow;
        if (gr >= G) gr = G - 1;
        const float* arow = srcs[which] + (size_t)gr * HIDC;
        v8f acc = {};
#pragma unroll
        for (int kf = 0; kf < 8; ++kf)
            acc = wmma_bf16(a_frag_f32(arow, kf, hl), b_frag(pk[which], 8, nt, kf, lane), acc);
#pragma unroll
        for (int r = 0; r < 8; ++r)
            raw[r + 8 * hl][j * 16 + mrow] = (__bf16)fmaxf(acc[r], 0.0f);
    }
    __syncthreads();

    // final: [16 x 1536] @ [1536 x 512]; 32 n-tiles, 4 per wave, 48 k-frags
    for (int nt = w; nt < 32; nt += 8) {
        v8f acc = {};
        const __bf16* arow = &raw[mrow][0];
        for (int kf = 0; kf < 48; ++kf)
            acc = wmma_bf16(a_frag_bf16(arow, kf, hl), b_frag(p_fin, 48, nt, kf, lane), acc);
#pragma unroll
        for (int r = 0; r < 8; ++r) {
            int gr = gbase + r + 8 * hl;
            if (gr < G)
                out[(size_t)gr * OUTD + nt * 16 + mrow] = acc[r];
        }
    }
}

// ---------------- host launcher ----------------

extern "C" void kernel_launch(void* const* d_in, const int* in_sizes, int n_in,
                              void* d_out, int out_size, void* d_ws, size_t ws_size,
                              hipStream_t stream) {
    const float* x    = (const float*)d_in[0];
    const int*   seg  = (const int*)d_in[1];
    const float* wm_sw1 = (const float*)d_in[3];
    const float* wm_sb1 = (const float*)d_in[4];
    const float* wm_sw2 = (const float*)d_in[5];
    const float* wm_sb2 = (const float*)d_in[6];
    const float* wm_vw1 = (const float*)d_in[7];
    const float* wm_vb1 = (const float*)d_in[8];
    const float* wm_vw2 = (const float*)d_in[9];
    const float* wm_vb2 = (const float*)d_in[10];
    const float* wm_cw  = (const float*)d_in[11];
    const float* ws_sw1 = (const float*)d_in[12];
    const float* ws_sb1 = (const float*)d_in[13];
    const float* ws_sw2 = (const float*)d_in[14];
    const float* ws_sb2 = (const float*)d_in[15];
    const float* ws_vw1 = (const float*)d_in[16];
    const float* ws_vb1 = (const float*)d_in[17];
    const float* ws_vw2 = (const float*)d_in[18];
    const float* ws_vb2 = (const float*)d_in[19];
    const float* ws_cw  = (const float*)d_in[20];
    const float* mx_cw  = (const float*)d_in[21];
    const float* fin_w  = (const float*)d_in[22];
    float* out = (float*)d_out;

    const int N = in_sizes[0] / NODE_DIM;
    const int G = out_size / OUTD;

    char* ws = (char*)d_ws;
    size_t off = 0;
    auto take = [&](size_t bytes) -> char* {
        char* p = ws + off;
        off = (off + bytes + 255) & ~(size_t)255;
        return p;
    };

    __bf16* p_wm_s1 = (__bf16*)take((size_t)256 * 256 * 2);
    __bf16* p_wm_s2 = (__bf16*)take((size_t)256 * 16 * 2);
    __bf16* p_wm_v1 = (__bf16*)take((size_t)256 * 256 * 2);
    __bf16* p_wm_v2 = (__bf16*)take((size_t)256 * 256 * 2);
    __bf16* p_ws_s1 = (__bf16*)take((size_t)256 * 256 * 2);
    __bf16* p_ws_s2 = (__bf16*)take((size_t)256 * 16 * 2);
    __bf16* p_ws_v1 = (__bf16*)take((size_t)256 * 256 * 2);
    __bf16* p_ws_v2 = (__bf16*)take((size_t)256 * 256 * 2);
    __bf16* p_cw    = (__bf16*)take((size_t)256 * 512 * 2);
    __bf16* p_cs    = (__bf16*)take((size_t)256 * 512 * 2);
    __bf16* p_cm    = (__bf16*)take((size_t)256 * 512 * 2);
    __bf16* p_fin   = (__bf16*)take((size_t)1536 * 512 * 2);
    float*  sc_wm   = (float*)take((size_t)N * NHEADS * 4);
    float*  sc_ws   = (float*)take((size_t)N * NHEADS * 4);
    __bf16* vT_wm   = (__bf16*)take((size_t)HIDC * N * 2);
    __bf16* vT_ws   = (__bf16*)take((size_t)HIDC * N * 2);
    int*    bounds  = (int*)take((size_t)(G + 1) * 4);
    float*  pgw     = (float*)take((size_t)G * HIDC * 4);
    float*  pgs     = (float*)take((size_t)G * HIDC * 4);
    float*  pgm     = (float*)take((size_t)G * HIDC * 4);
    (void)ws_size; (void)n_in;

    // 0) weight repack into WMMA B-fragment layout
    k_pack_b<<<256, 256, 0, stream>>>(wm_sw1, p_wm_s1, 256, 256, 256);
    k_pack_b<<<32,  256, 0, stream>>>(wm_sw2, p_wm_s2, 256, 16, 8);
    k_pack_b<<<256, 256, 0, stream>>>(wm_vw1, p_wm_v1, 256, 256, 256);
    k_pack_b<<<256, 256, 0, stream>>>(wm_vw2, p_wm_v2, 256, 256, 256);
    k_pack_b<<<256, 256, 0, stream>>>(ws_sw1, p_ws_s1, 256, 256, 256);
    k_pack_b<<<32,  256, 0, stream>>>(ws_sw2, p_ws_s2, 256, 16, 8);
    k_pack_b<<<256, 256, 0, stream>>>(ws_vw1, p_ws_v1, 256, 256, 256);
    k_pack_b<<<256, 256, 0, stream>>>(ws_vw2, p_ws_v2, 256, 256, 256);
    k_pack_b<<<256, 256, 0, stream>>>(wm_cw,  p_cw,  256, 512, 512);
    k_pack_b<<<256, 256, 0, stream>>>(ws_cw,  p_cs,  256, 512, 512);
    k_pack_b<<<256, 256, 0, stream>>>(mx_cw,  p_cm,  256, 512, 512);
    k_pack_b<<<512, 256, 0, stream>>>(fin_w,  p_fin, 1536, 512, 512);

    // 1) segment bounds
    k_bounds<<<(G + 1 + 255) / 256, 256, 0, stream>>>(seg, N, G, bounds);

    // 2) fused per-node MLPs (WMMA), 32 nodes per wave, 4 waves per block
    int tiles32 = (N + 31) / 32;
    k_node_mlps<<<(tiles32 + 3) / 4, 128, 0, stream>>>(
        x, N,
        p_wm_s1, wm_sb1, p_wm_s2, wm_sb2, p_wm_v1, wm_vb1, p_wm_v2, wm_vb2,
        p_ws_s1, ws_sb1, p_ws_s2, ws_sb2, p_ws_v1, ws_vb1, p_ws_v2, ws_vb2,
        sc_wm, sc_ws, vT_wm, vT_ws);

    // 3) per-graph softmax + pools
    k_segment_pool<<<G, 256, 0, stream>>>(x, bounds, N, sc_wm, sc_ws, vT_wm, vT_ws,
                                          pgw, pgs, pgm);

    // 4) per-graph output GEMMs (WMMA)
    k_graph_out<<<(G + 15) / 16, 256, 0, stream>>>(pgw, pgs, pgm, p_cw, p_cs, p_cm,
                                                   p_fin, out, G);
}
